// MLA_22239340659163
// MI455X (gfx1250) — matrix operand order repaired and checked by reading, not compiled, not run on hardware
//
#include <hip/hip_runtime.h>
#include <math.h>

// ---------------- problem constants ----------------
#define B_    2
#define S_    2048
#define D_    2048
#define NH_   16
#define DH_   128
#define DR_   64
#define DCQ_  1536
#define DCKV_ 512
#define DQK_  192          // DH + DR
#define MROWS (B_ * S_)    // 4096

typedef __bf16 bf16;
typedef __attribute__((ext_vector_type(16))) __bf16 v16bf;
typedef __attribute__((ext_vector_type(8)))  __bf16 v8bf;
typedef __attribute__((ext_vector_type(8)))  float  v8f;
typedef __attribute__((ext_vector_type(4)))  int    v4i_;

// ---------------- async global->LDS copy path (gfx1250 ASYNCcnt) ----------------
#if defined(__AMDGCN__) && __has_builtin(__builtin_amdgcn_global_load_async_to_lds_b128)
#define ASYNC_COPY 1
typedef __attribute__((address_space(1))) v4i_* as1_v4i_p;   // global v4i*
typedef __attribute__((address_space(3))) v4i_* as3_v4i_p;   // LDS v4i*
#else
#define ASYNC_COPY 0
#endif

__device__ __forceinline__ void copy16_to_lds(const bf16* g, bf16* l) {
#if ASYNC_COPY
    __builtin_amdgcn_global_load_async_to_lds_b128(
        (as1_v4i_p)(v4i_*)(void*)const_cast<bf16*>(g),
        (as3_v4i_p)(v4i_*)(void*)l,
        0, 0);
#else
    *(v8bf*)l = *(const v8bf*)g;
#endif
}

__device__ __forceinline__ void async_join() {
#if ASYNC_COPY
#if __has_builtin(__builtin_amdgcn_s_wait_asynccnt)
    __builtin_amdgcn_s_wait_asynccnt(0);
#else
    asm volatile("s_wait_asynccnt 0x0" ::: "memory");
#endif
#endif
}

// ---------------- WMMA fragment helpers (wave32, 16x16x32 bf16) ----------------
// A (16x32, MxK) layout: lane m=L%16, g=L/16 holds K = [k+g*8 .. +7] and [k+16+g*8 .. +7]
__device__ __forceinline__ v16bf load_frag_a(const bf16* base, int ld, int row, int k) {
    const int lane = threadIdx.x & 31;
    const int m = lane & 15, g = lane >> 4;
    const bf16* p = base + (size_t)(row + m) * ld + k + g * 8;
    v8bf lo = *(const v8bf*)p;
    v8bf hi = *(const v8bf*)(p + 16);
    return __builtin_shufflevector(lo, hi, 0, 1, 2, 3, 4, 5, 6, 7,
                                           8, 9, 10, 11, 12, 13, 14, 15);
}

// B (32x16, KxN) layout: lane n=L%16, g=L/16 holds K = [k+g*8 .. +7] and [k+16+g*8 .. +7] of column n.
// Bt is stored N x K row-major so the 16 elements are two contiguous 16B loads.
__device__ __forceinline__ v16bf load_frag_b(const bf16* baseT, int ld, int col, int k) {
    const int lane = threadIdx.x & 31;
    const int n = lane & 15, g = lane >> 4;
    const bf16* p = baseT + (size_t)(col + n) * ld + k + g * 8;
    v8bf lo = *(const v8bf*)p;
    v8bf hi = *(const v8bf*)(p + 16);
    return __builtin_shufflevector(lo, hi, 0, 1, 2, 3, 4, 5, 6, 7,
                                           8, 9, 10, 11, 12, 13, 14, 15);
}

__device__ __forceinline__ v8f wmma_bf16(v16bf a, v16bf b, v8f c) {
    return __builtin_amdgcn_wmma_f32_16x16x32_bf16(false, a, false, b, (short)0, c, false, false);
}

// by-value accumulator store: keeps accumulators in SSA (no address-taken allocas)
__device__ __forceinline__ void store_tile(v8f acc, float* __restrict__ C,
                                           const float* __restrict__ resid,
                                           int N, int mt, int nt) {
    #pragma unroll
    for (int r = 0; r < 8; ++r) {
        size_t idx = (size_t)(mt + r) * N + nt;
        float v = acc[r];
        if (resid) v += resid[idx];
        C[idx] = v;
    }
}

// ---------------- generic GEMM: C[MxN] = A[MxK](bf16) * Bt[NxK](bf16) (+resid) ----------------
// block = 256 thr = 8 waves laid out 4(M) x 2(N); wave tile 32x32; block tile 128x64x32.
// A/B block tiles double-buffered in LDS, filled with async global->LDS copies.
#define BM 128
#define BN 64
#define BK 32
__global__ __launch_bounds__(256)
void mla_gemm_bf16_kernel(const bf16* __restrict__ A, const bf16* __restrict__ Bt,
                          float* __restrict__ C, const float* __restrict__ resid,
                          int M, int N, int K) {
    __shared__ bf16 sA[2][BM * BK];   // 2 x 8KB
    __shared__ bf16 sB[2][BN * BK];   // 2 x 4KB
    const int lane = threadIdx.x & 31;
    const int wave = threadIdx.x >> 5;
    const int wm = wave & 3, wn = wave >> 2;
    const int mBlk = blockIdx.y * BM;
    const int nBlk = blockIdx.x * BN;
    const int t = threadIdx.x;

    // cooperative tile stage: A = 512 16B chunks (2/thread), B = 256 chunks (1/thread)
    auto stage = [&](int buf, int k) {
        int c = t;
        #pragma unroll
        for (int it = 0; it < 2; ++it, c += 256) {
            const int row = c >> 2, col = (c & 3) * 8;
            copy16_to_lds(A + (size_t)(mBlk + row) * K + k + col, &sA[buf][row * BK + col]);
        }
        {
            const int row = t >> 2, col = (t & 3) * 8;
            copy16_to_lds(Bt + (size_t)(nBlk + row) * K + k + col, &sB[buf][row * BK + col]);
        }
    };

    v8f acc00 = {}, acc01 = {}, acc10 = {}, acc11 = {};
    stage(0, 0);
    async_join();
    __syncthreads();

    int buf = 0;
    for (int k = 0; k < K; k += BK) {
        if (k + BK < K) stage(buf ^ 1, k + BK);   // prefetch next tile (overlaps compute)
        v16bf a0 = load_frag_a(sA[buf], BK, wm * 32,      0);
        v16bf a1 = load_frag_a(sA[buf], BK, wm * 32 + 16, 0);
        v16bf b0 = load_frag_b(sB[buf], BK, wn * 32,      0);
        v16bf b1 = load_frag_b(sB[buf], BK, wn * 32 + 16, 0);
        acc00 = wmma_bf16(a0, b0, acc00);
        acc01 = wmma_bf16(a0, b1, acc01);
        acc10 = wmma_bf16(a1, b0, acc10);
        acc11 = wmma_bf16(a1, b1, acc11);
        async_join();
        __syncthreads();
        buf ^= 1;
    }

    const int mBase = mBlk + wm * 32;
    const int nBase = nBlk + wn * 32;
    const int n = lane & 15, g = lane >> 4;
    store_tile(acc00, C, resid, N, mBase + g * 8,      nBase + n);
    store_tile(acc01, C, resid, N, mBase + g * 8,      nBase + 16 + n);
    store_tile(acc10, C, resid, N, mBase + 16 + g * 8, nBase + n);
    store_tile(acc11, C, resid, N, mBase + 16 + g * 8, nBase + 16 + n);
}

// ---------------- RMSNorm: Y(bf16) = X * rsqrt(mean(X^2)+eps) * gain ----------------
__global__ __launch_bounds__(256)
void mla_rmsnorm_kernel(const float* __restrict__ X, const float* __restrict__ gain,
                        bf16* __restrict__ Y, int N) {
    const int row = blockIdx.x;
    const float* x = X + (size_t)row * N;
    bf16* y = Y + (size_t)row * N;
    __shared__ float red[256];
    float ss = 0.f;
    for (int i = threadIdx.x; i < N; i += 256) { float v = x[i]; ss += v * v; }
    red[threadIdx.x] = ss;
    __syncthreads();
    for (int s = 128; s > 0; s >>= 1) {
        if (threadIdx.x < s) red[threadIdx.x] += red[threadIdx.x + s];
        __syncthreads();
    }
    const float scale = rsqrtf(red[0] / (float)N + 1.1920929e-07f);
    for (int i = threadIdx.x; i < N; i += 256) y[i] = (bf16)(x[i] * scale * gain[i]);
}

// ---------------- weight transpose + fp32->bf16: W[KxN] -> Wt[NxK] ----------------
__global__ __launch_bounds__(256)
void mla_transpose_conv_kernel(const float* __restrict__ W, bf16* __restrict__ Wt,
                               int K, int N) {
    __shared__ float tile[16][17];
    const int kb = blockIdx.y * 16, nb = blockIdx.x * 16;
    const int tx = threadIdx.x & 15, ty = threadIdx.x >> 4;
    tile[ty][tx] = W[(size_t)(kb + ty) * N + nb + tx];
    __syncthreads();
    Wt[(size_t)(nb + ty) * K + kb + tx] = (bf16)tile[tx][ty];
}

// ---------------- Q pack: concat(q_c, rope(q_R)) * 1/sqrt(192) -> (b,h,s,192) bf16 ----------
__global__ __launch_bounds__(256)
void mla_q_pack_kernel(const float* __restrict__ Qc, const float* __restrict__ QR,
                       const float* __restrict__ cosT, const float* __restrict__ sinT,
                       bf16* __restrict__ Qb) {
    const int row = blockIdx.x;             // b*S + s
    const int b = row >> 11, s = row & (S_ - 1);
    const float scale = 0.07216878364870323f;   // 1/sqrt(192)
    for (int t = threadIdx.x; t < NH_ * DQK_; t += 256) {
        const int h = t / DQK_, d = t % DQK_;
        float v;
        if (d < DH_) {
            v = Qc[(size_t)row * (NH_ * DH_) + h * DH_ + d];
        } else {
            const int dr = d - DH_, i = dr >> 1;
            const float x1 = QR[(size_t)row * (NH_ * DR_) + h * DR_ + 2 * i];
            const float x2 = QR[(size_t)row * (NH_ * DR_) + h * DR_ + 2 * i + 1];
            const float c = cosT[s * (DR_ / 2) + i], sn = sinT[s * (DR_ / 2) + i];
            v = (dr & 1) ? (x1 * sn + x2 * c) : (x1 * c - x2 * sn);
        }
        Qb[((size_t)(b * NH_ + h) * S_ + s) * DQK_ + d] = (bf16)(v * scale);
    }
}

// ---------------- K pack: concat(k_c, broadcast rope(k_R)) -> (b,h,s,192) bf16 --------------
__global__ __launch_bounds__(256)
void mla_k_pack_kernel(const float* __restrict__ Kc, const float* __restrict__ kR,
                       const float* __restrict__ cosT, const float* __restrict__ sinT,
                       bf16* __restrict__ Kb) {
    const int row = blockIdx.x;
    const int b = row >> 11, s = row & (S_ - 1);
    for (int t = threadIdx.x; t < NH_ * DQK_; t += 256) {
        const int h = t / DQK_, d = t % DQK_;
        float v;
        if (d < DH_) {
            v = Kc[(size_t)row * (NH_ * DH_) + h * DH_ + d];
        } else {
            const int dr = d - DH_, i = dr >> 1;
            const float x1 = kR[(size_t)row * DR_ + 2 * i];
            const float x2 = kR[(size_t)row * DR_ + 2 * i + 1];
            const float c = cosT[s * (DR_ / 2) + i], sn = sinT[s * (DR_ / 2) + i];
            v = (dr & 1) ? (x1 * sn + x2 * c) : (x1 * c - x2 * sn);
        }
        Kb[((size_t)(b * NH_ + h) * S_ + s) * DQK_ + d] = (bf16)v;
    }
}

// ---------------- V pack: (b,s,h*128+d) f32 -> Vt (b,h,128,s) bf16 -------------------------
__global__ __launch_bounds__(256)
void mla_v_pack_kernel(const float* __restrict__ Vraw, bf16* __restrict__ Vt) {
    const int row = blockIdx.x;
    const int b = row >> 11, s = row & (S_ - 1);
    for (int t = threadIdx.x; t < NH_ * DH_; t += 256) {
        const int h = t >> 7, d = t & 127;
        Vt[((size_t)(b * NH_ + h) * DH_ + d) * S_ + s] = (bf16)Vraw[(size_t)row * (NH_ * DH_) + t];
    }
}

// ---------------- attention: per (b,h,16-query tile), scores in LDS, softmax, P*V ----------
__global__ __launch_bounds__(256)
void mla_attn_kernel(const bf16* __restrict__ Q, const bf16* __restrict__ Kb,
                     const bf16* __restrict__ Vt, bf16* __restrict__ O) {
    extern __shared__ float smem[];
    float* sc     = smem;                    // [16][2048] scores / probs
    float* red    = smem + 16 * S_;          // [16][16] partial reductions
    float* rowmax = red + 256;               // [16]
    float* rowinv = rowmax + 16;             // [16]

    const int bid = blockIdx.x;
    const int qt = bid & 127;                // S/16 = 128 query tiles
    const int h  = (bid >> 7) & 15;
    const int b  = bid >> 11;
    const int s0 = qt * 16;
    const int lane = threadIdx.x & 31;
    const int wave = threadIdx.x >> 5;

    const bf16* qbase = Q  + (size_t)(b * NH_ + h) * S_ * DQK_;
    const bf16* kbase = Kb + (size_t)(b * NH_ + h) * S_ * DQK_;
    const bf16* vbase = Vt + (size_t)(b * NH_ + h) * DH_ * S_;

    // Q fragments for this tile (all 192 of K-dim), kept in registers
    v16bf qf0 = load_frag_a(qbase, DQK_, s0, 0);
    v16bf qf1 = load_frag_a(qbase, DQK_, s0, 32);
    v16bf qf2 = load_frag_a(qbase, DQK_, s0, 64);
    v16bf qf3 = load_frag_a(qbase, DQK_, s0, 96);
    v16bf qf4 = load_frag_a(qbase, DQK_, s0, 128);
    v16bf qf5 = load_frag_a(qbase, DQK_, s0, 160);

    // ---- scores: only key tiles kt <= qt (rest are fully causal-masked) ----
    {
        const int n = lane & 15, g = lane >> 4;
        for (int kt = wave; kt <= qt; kt += 8) {
            v8f acc = {};
            acc = wmma_bf16(qf0, load_frag_b(kbase, DQK_, kt * 16, 0),   acc);
            acc = wmma_bf16(qf1, load_frag_b(kbase, DQK_, kt * 16, 32),  acc);
            acc = wmma_bf16(qf2, load_frag_b(kbase, DQK_, kt * 16, 64),  acc);
            acc = wmma_bf16(qf3, load_frag_b(kbase, DQK_, kt * 16, 96),  acc);
            acc = wmma_bf16(qf4, load_frag_b(kbase, DQK_, kt * 16, 128), acc);
            acc = wmma_bf16(qf5, load_frag_b(kbase, DQK_, kt * 16, 160), acc);
            #pragma unroll
            for (int r = 0; r < 8; ++r)
                sc[(g * 8 + r) * S_ + kt * 16 + n] = acc[r];
        }
    }
    __syncthreads();

    // ---- causal mask + softmax (16 row-teams of 16 threads) ----
    const int r  = threadIdx.x >> 4;   // query row in tile
    const int c0 = threadIdx.x & 15;
    const int qAbs = s0 + r;
    float mx = -INFINITY;
    for (int kk = c0; kk < S_; kk += 16) {
        float v = sc[r * S_ + kk];
        if (kk > qAbs) { v = -INFINITY; sc[r * S_ + kk] = v; }
        mx = fmaxf(mx, v);
    }
    red[r * 16 + c0] = mx;
    __syncthreads();
    if (c0 == 0) {
        float m = -INFINITY;
        for (int j = 0; j < 16; ++j) m = fmaxf(m, red[r * 16 + j]);
        rowmax[r] = m;
    }
    __syncthreads();
    const float m = rowmax[r];
    float sum = 0.f;
    for (int kk = c0; kk < S_; kk += 16) {
        float e = __expf(sc[r * S_ + kk] - m);
        sc[r * S_ + kk] = e;
        sum += e;
    }
    red[r * 16 + c0] = sum;
    __syncthreads();
    if (c0 == 0) {
        float s = 0.f;
        for (int j = 0; j < 16; ++j) s += red[r * 16 + j];
        rowinv[r] = 1.0f / s;
    }
    __syncthreads();

    // ---- out = P * V : wave w owns output columns [w*16, w*16+16) of DH=128 ----
    {
        const int mL = lane & 15, g = lane >> 4;
        const float inv = rowinv[mL];
        v8f oacc = {};
        for (int kb2 = 0; kb2 < S_; kb2 += 32) {
            const float* prow = &sc[mL * S_ + kb2 + g * 8];
            v16bf pv;
            #pragma unroll
            for (int j = 0; j < 8; ++j) pv[j]     = (bf16)(prow[j] * inv);
            #pragma unroll
            for (int j = 0; j < 8; ++j) pv[j + 8] = (bf16)(prow[16 + j] * inv);
            v16bf vb = load_frag_b(vbase, S_, wave * 16, kb2);
            oacc = wmma_bf16(pv, vb, oacc);
        }
        const int n = lane & 15;
        #pragma unroll
        for (int rr = 0; rr < 8; ++rr) {
            const int sRow = s0 + g * 8 + rr;
            O[((size_t)(b * S_ + sRow)) * (NH_ * DH_) + h * DH_ + wave * 16 + n] = (bf16)oacc[rr];
        }
    }
}

// ---------------- host orchestration ----------------
extern "C" void kernel_launch(void* const* d_in, const int* in_sizes, int n_in,
                              void* d_out, int out_size, void* d_ws, size_t ws_size,
                              hipStream_t stream) {
    const float* x     = (const float*)d_in[0];
    // d_in[1] = mask (causal, known statically) - unused
    const float* cosT  = (const float*)d_in[2];
    const float* sinT  = (const float*)d_in[3];
    const float* g_in  = (const float*)d_in[4];
    const float* g_cq  = (const float*)d_in[5];
    const float* g_ckv = (const float*)d_in[6];
    const float* W_cq  = (const float*)d_in[7];
    const float* W_q   = (const float*)d_in[8];
    const float* W_qR  = (const float*)d_in[9];
    const float* W_ckv = (const float*)d_in[10];
    const float* W_k   = (const float*)d_in[11];
    const float* W_kR  = (const float*)d_in[12];
    const float* W_v   = (const float*)d_in[13];
    const float* W_o   = (const float*)d_in[14];
    (void)in_sizes; (void)n_in; (void)out_size; (void)ws_size;

    char* ws = (char*)d_ws;
    size_t off = 0;
    auto alloc = [&](size_t bytes) -> char* {
        char* p = ws + off;
        off = (off + bytes + 255) & ~(size_t)255;
        return p;
    };

    const int M = MROWS;
    bf16*  H      = (bf16*)alloc((size_t)M * D_ * 2);
    bf16*  Cq     = (bf16*)alloc((size_t)M * DCQ_ * 2);
    bf16*  Ckv    = (bf16*)alloc((size_t)M * DCKV_ * 2);
    float* kRraw  = (float*)alloc((size_t)M * DR_ * 4);
    bf16*  Qb     = (bf16*)alloc((size_t)B_ * NH_ * S_ * DQK_ * 2);
    bf16*  Kb     = (bf16*)alloc((size_t)B_ * NH_ * S_ * DQK_ * 2);
    bf16*  Vt     = (bf16*)alloc((size_t)B_ * NH_ * DH_ * S_ * 2);
    bf16*  Obuf   = (bf16*)alloc((size_t)M * D_ * 2);
    bf16*  Wt_cq  = (bf16*)alloc((size_t)DCQ_ * D_ * 2);
    bf16*  Wt_q   = (bf16*)alloc((size_t)(NH_ * DH_) * DCQ_ * 2);
    bf16*  Wt_qR  = (bf16*)alloc((size_t)(NH_ * DR_) * DCQ_ * 2);
    bf16*  Wt_ckv = (bf16*)alloc((size_t)DCKV_ * D_ * 2);
    bf16*  Wt_k   = (bf16*)alloc((size_t)(NH_ * DH_) * DCKV_ * 2);
    bf16*  Wt_kR  = (bf16*)alloc((size_t)DR_ * D_ * 2);
    bf16*  Wt_v   = (bf16*)alloc((size_t)D_ * DCKV_ * 2);
    bf16*  Wt_o   = (bf16*)alloc((size_t)D_ * (NH_ * DH_) * 2);
    float* F1     = (float*)alloc((size_t)M * D_ * 4);            // reused f32 scratch
    float* F2     = (float*)alloc((size_t)M * (NH_ * DR_) * 4);   // second f32 scratch

    auto tr = [&](const float* W, bf16* Wt, int K, int N) {
        dim3 g(N / 16, K / 16);
        mla_transpose_conv_kernel<<<g, 256, 0, stream>>>(W, Wt, K, N);
    };
    auto gemm = [&](const bf16* A, const bf16* Bt, float* C, const float* resid,
                    int Mm, int Nn, int Kk) {
        dim3 g(Nn / 64, Mm / 128);
        mla_gemm_bf16_kernel<<<g, 256, 0, stream>>>(A, Bt, C, resid, Mm, Nn, Kk);
    };

    // 1) weights -> bf16 N x K
    tr(W_cq,  Wt_cq,  D_,       DCQ_);
    tr(W_q,   Wt_q,   DCQ_,     NH_ * DH_);
    tr(W_qR,  Wt_qR,  DCQ_,     NH_ * DR_);
    tr(W_ckv, Wt_ckv, D_,       DCKV_);
    tr(W_k,   Wt_k,   DCKV_,    NH_ * DH_);
    tr(W_kR,  Wt_kR,  D_,       DR_);
    tr(W_v,   Wt_v,   DCKV_,    D_);
    tr(W_o,   Wt_o,   NH_ * DH_, D_);

    // 2) h = rms(x)*g_in
    mla_rmsnorm_kernel<<<M, 256, 0, stream>>>(x, g_in, H, D_);

    // 3) latents
    gemm(H, Wt_cq, F1, nullptr, M, DCQ_, D_);
    mla_rmsnorm_kernel<<<M, 256, 0, stream>>>(F1, g_cq, Cq, DCQ_);
    gemm(H, Wt_ckv, F1, nullptr, M, DCKV_, D_);
    mla_rmsnorm_kernel<<<M, 256, 0, stream>>>(F1, g_ckv, Ckv, DCKV_);
    gemm(H, Wt_kR, kRraw, nullptr, M, DR_, D_);

    // 4) Q
    gemm(Cq, Wt_q,  F1, nullptr, M, NH_ * DH_, DCQ_);
    gemm(Cq, Wt_qR, F2, nullptr, M, NH_ * DR_, DCQ_);
    mla_q_pack_kernel<<<M, 256, 0, stream>>>(F1, F2, cosT, sinT, Qb);

    // 5) K
    gemm(Ckv, Wt_k, F1, nullptr, M, NH_ * DH_, DCKV_);
    mla_k_pack_kernel<<<M, 256, 0, stream>>>(F1, kRraw, cosT, sinT, Kb);

    // 6) V (transposed)
    gemm(Ckv, Wt_v, F1, nullptr, M, D_, DCKV_);
    mla_v_pack_kernel<<<M, 256, 0, stream>>>(F1, Vt);

    // 7) attention (128KB + eps dynamic LDS per block; gfx1250 WGP has 320KB)
    const size_t ldsBytes = (size_t)(16 * S_ + 256 + 32) * sizeof(float);
    (void)hipFuncSetAttribute((const void*)mla_attn_kernel,
                              hipFuncAttributeMaxDynamicSharedMemorySize, (int)ldsBytes);
    mla_attn_kernel<<<B_ * NH_ * (S_ / 16), 256, ldsBytes, stream>>>(Qb, Kb, Vt, Obuf);

    // 8) out = x + O @ W_o
    gemm(Obuf, Wt_o, (float*)d_out, x, M, D_, NH_ * DH_);
}
